// MultiHeadedAttention_29222957482433
// MI455X (gfx1250) — matrix operand order repaired and checked by reading, not compiled, not run on hardware
//
#include <hip/hip_runtime.h>
#include <hip/hip_bf16.h>

// ---------------------------------------------------------------------------
// Intersample multi-head attention for MI455X (gfx1250), wave32 + WMMA bf16
// + async global->LDS staging (ASYNCcnt path).
//   D_MODEL=1024, H=16, D_K=64, B=128 (attention dim), N=256
// Pipeline:
//   k_cvt      : fp32 -> bf16 (weights + activations) into ws
//   k_proj     : Y = X*W^T + b  (WMMA bf16, double-buffered async A staging),
//                scatter to q/k: [h][n][b][d]   v: [h][n][d][b]
//   k_attn     : per (h,n): S=Q*K^T/8, softmax over batch, O=P*V (WMMA)
//   k_outproj  : Z = O*Wo^T + bo  -> fp32 d_out
// ---------------------------------------------------------------------------

#define D_MODEL 1024
#define NHEAD   16
#define DK      64
#define BATCH   128
#define NSAMP   256
#define M_TOT   (BATCH * NSAMP)   // 32768

typedef __attribute__((ext_vector_type(4)))  int    v4i;
typedef __attribute__((ext_vector_type(4)))  __bf16 v4bf;
typedef __attribute__((ext_vector_type(8)))  __bf16 v8bf;
typedef __attribute__((ext_vector_type(16))) __bf16 v16bf;
typedef __attribute__((ext_vector_type(8)))  float  v8f;

#if defined(__AMDGCN__) && __has_builtin(__builtin_amdgcn_global_load_async_to_lds_b128)
#define HAS_ASYNC_CP 1
#else
#define HAS_ASYNC_CP 0
#endif
#if defined(__AMDGCN__) && !HAS_ASYNC_CP
#warning "global_load_async_to_lds_b128 builtin unavailable; using inline asm"
#endif

// 16B async copy global -> LDS (tracked by ASYNCcnt).
// Builtin signature (from toolchain diagnostics): (AS1 v4i*, AS3 v4i*, Ii, Ii)
__device__ __forceinline__ void async_cp16(void* lds, const void* gsrc) {
#if HAS_ASYNC_CP
  __builtin_amdgcn_global_load_async_to_lds_b128(
      (__attribute__((address_space(1))) v4i*)gsrc,
      (__attribute__((address_space(3))) v4i*)lds, 0, 0);
#else
  unsigned loff =
      (unsigned)(unsigned long long)(__attribute__((address_space(3))) void*)lds;
  asm volatile("global_load_async_to_lds_b128 %0, %1, off"
               :: "v"(loff), "v"(gsrc)
               : "memory");
#endif
}

__device__ __forceinline__ void wait_async0() {
#if defined(__AMDGCN__) && __has_builtin(__builtin_amdgcn_s_wait_asynccnt)
  __builtin_amdgcn_s_wait_asynccnt(0);
#else
  asm volatile("s_wait_asynccnt 0x0" ::: "memory");
#endif
}

// Load a 16-element bf16 fragment as two 16B halves (elements 0..7 from p0,
// 8..15 from p1). Matches ISA A/B per-lane element ordering.
__device__ __forceinline__ v16bf frag_ld(const __bf16* p0, const __bf16* p1) {
  v8bf lo = *(const v8bf*)p0;
  v8bf hi = *(const v8bf*)p1;
  v16bf r;
#pragma unroll
  for (int i = 0; i < 8; ++i) { r[i] = lo[i]; r[i + 8] = hi[i]; }
  return r;
}

__device__ __forceinline__ v8f wmma_bf16(v16bf a, v16bf b, v8f c) {
  return __builtin_amdgcn_wmma_f32_16x16x32_bf16(
      /*neg_a=*/false, a, /*neg_b=*/false, b,
      /*c_mod=*/(short)0, c, /*reuse_a=*/false, /*reuse_b=*/false);
}

// ---------------------------------------------------------------------------
// fp32 -> bf16 conversion, 4 elems/thread
// ---------------------------------------------------------------------------
__global__ void __launch_bounds__(256) k_cvt(const float* __restrict__ in,
                                             __bf16* __restrict__ out, int n4) {
  int i = blockIdx.x * blockDim.x + threadIdx.x;
  if (i >= n4) return;
  float4 f = ((const float4*)in)[i];
  v4bf o;
  o[0] = (__bf16)f.x; o[1] = (__bf16)f.y; o[2] = (__bf16)f.z; o[3] = (__bf16)f.w;
  ((v4bf*)out)[i] = o;
}

// ---------------------------------------------------------------------------
// Projection GEMM: Y[m][e] = sum_d X[m][d]*W[e][d] + bias[e]
// M=32768, E=1024, K=1024, all-bf16 operands. Block tile 128x128,
// 8 waves (2M x 4E), wave tile 64x32 (4x2 WMMA tiles).
// A tile (128x32 bf16 = 8KB) staged into double-buffered LDS with
// global_load_async_to_lds_b128 in WMMA-A-fragment order; B fragments read
// straight from bf16 W (row-major, two 8-K runs per lane, 16 apart).
// vtrans=0: out[((h*N+n)*B + b)*DK + d]   (q,k layout)
// vtrans=1: out[((h*N+n)*DK + d)*B + b]   (v layout, pre-transposed for PV)
// ---------------------------------------------------------------------------
#define ASLOT 24   // 16 payload bf16 + 8 pad -> 48B slot stride (bank spread)

__global__ void __launch_bounds__(256)
k_proj(const __bf16* __restrict__ X, const __bf16* __restrict__ W,
       const float* __restrict__ bias, __bf16* __restrict__ out, int vtrans) {
  __shared__ __attribute__((aligned(32))) __bf16 Alds[2][8][32][ASLOT];  // 24KB

  const int tid  = threadIdx.x;
  const int lane = tid & 31, wave = tid >> 5;
  const int wm = wave & 1, we = wave >> 1;      // 2 x 4 wave grid
  const int half = lane >> 4, l16 = lane & 15;
  const int mBlk = blockIdx.x * 128;
  const int eBlk = blockIdx.y * 128;

  // async staging: 512 x 16B chunks, 2 per thread.
  // chunk c: row m = c>>2, K-group g = (c>>1)&1, 8-elem half h = c&1
  // dest slot: Alds[buf][m>>4][h*16 + (m&15)][g*8]  (fragment order)
  auto stage = [&](int buf, int k0) {
#pragma unroll
    for (int i = 0; i < 2; ++i) {
      const int c = tid * 2 + i;
      const int m = c >> 2, g = (c >> 1) & 1, h = c & 1;
      const __bf16* src = X + (size_t)(mBlk + m) * D_MODEL + k0 + g * 16 + h * 8;
      __bf16* dst = &Alds[buf][m >> 4][h * 16 + (m & 15)][g * 8];
      async_cp16(dst, src);
    }
  };

  v8f acc[4][2];
#pragma unroll
  for (int mt = 0; mt < 4; ++mt)
#pragma unroll
    for (int et = 0; et < 2; ++et) acc[mt][et] = (v8f){};

  stage(0, 0);
  for (int kk = 0; kk < D_MODEL / 32; ++kk) {
    const int k0 = kk * 32;
    const int buf = kk & 1;
    wait_async0();        // our copies for `buf` have landed in LDS
    __syncthreads();      // everyone's copies landed; prev reads of buf^1 done
    if (kk + 1 < D_MODEL / 32) stage(buf ^ 1, k0 + 32);  // overlap next tile

    v16bf bfrag[2];
#pragma unroll
    for (int et = 0; et < 2; ++et) {
      const int e = eBlk + we * 32 + et * 16 + l16;
      const __bf16* bp = W + (size_t)e * D_MODEL + k0 + half * 8;
      bfrag[et] = frag_ld(bp, bp + 16);
    }
#pragma unroll
    for (int mt = 0; mt < 4; ++mt) {
      const __bf16* ap = &Alds[buf][wm * 4 + mt][lane][0];
      v16bf afrag = frag_ld(ap, ap + 8);
      acc[mt][0] = wmma_bf16(afrag, bfrag[0], acc[mt][0]);
      acc[mt][1] = wmma_bf16(afrag, bfrag[1], acc[mt][1]);
    }
  }

  // epilogue: bias + bf16 scatter into attention layout
#pragma unroll
  for (int et = 0; et < 2; ++et) {
    const int e  = eBlk + we * 32 + et * 16 + l16;
    const float bv = bias[e];
    const int h = e >> 6, d = e & 63;
#pragma unroll
    for (int mt = 0; mt < 4; ++mt) {
#pragma unroll
      for (int r = 0; r < 8; ++r) {
        const int m = mBlk + wm * 64 + mt * 16 + r + half * 8;
        const int b = m >> 8, n = m & 255;
        const float val = acc[mt][et][r] + bv;
        const size_t o = vtrans
            ? ((size_t)(h * NSAMP + n) * DK + d) * BATCH + b
            : ((size_t)(h * NSAMP + n) * BATCH + b) * DK + d;
        out[o] = (__bf16)val;
      }
    }
  }
}

// ---------------------------------------------------------------------------
// Attention: one block per (h,n). q,k: [b][d] (128x64), v: [d][b] transposed.
// Each wave owns 16 rows (b) of the 128x128 score matrix.
// ---------------------------------------------------------------------------
#define PROW 136   // 128 payload + 8 pad -> 68-dword row stride (bank spread)

__global__ void __launch_bounds__(256)
k_attn(const __bf16* __restrict__ q, const __bf16* __restrict__ k,
       const __bf16* __restrict__ v, __bf16* __restrict__ out) {
  __shared__ __attribute__((aligned(32))) __bf16 P[8][16][PROW];  // 34KB

  const int hn = blockIdx.x;               // h*256 + n
  const int h = hn >> 8, n = hn & 255;
  const int lane = threadIdx.x & 31, wave = threadIdx.x >> 5;
  const int half = lane >> 4, l16 = lane & 15;

  const __bf16* qb = q + (size_t)hn * (BATCH * DK);
  const __bf16* kb = k + (size_t)hn * (BATCH * DK);
  const __bf16* vb = v + (size_t)hn * (DK * BATCH);

  // ---- S = Q * K^T : wave rows [wave*16, +16), 8 column tiles ----
  v8f s[8];
#pragma unroll
  for (int t = 0; t < 8; ++t) s[t] = (v8f){};

#pragma unroll
  for (int ks = 0; ks < 2; ++ks) {         // contraction dim d, 2 steps of 32
    const __bf16* qp = qb + (size_t)(wave * 16 + l16) * DK + ks * 32 + half * 8;
    v16bf qf = frag_ld(qp, qp + 16);
#pragma unroll
    for (int t = 0; t < 8; ++t) {
      const __bf16* kp = kb + (size_t)(t * 16 + l16) * DK + ks * 32 + half * 8;
      v16bf kf = frag_ld(kp, kp + 16);
      s[t] = wmma_bf16(qf, kf, s[t]);
    }
  }

  // ---- softmax over batch dim (rows spread over 8 tiles + 16 lanes) ----
  const float scale = 0.125f;              // 1/sqrt(64)
#pragma unroll
  for (int r = 0; r < 8; ++r) {
    float mx = -3.402823e38f;
#pragma unroll
    for (int t = 0; t < 8; ++t) mx = fmaxf(mx, s[t][r]);
#pragma unroll
    for (int msk = 1; msk <= 8; msk <<= 1)
      mx = fmaxf(mx, __shfl_xor(mx, msk, 32));
    float sum = 0.0f;
#pragma unroll
    for (int t = 0; t < 8; ++t) {
      float p = __expf((s[t][r] - mx) * scale);
      s[t][r] = p;
      sum += p;
    }
#pragma unroll
    for (int msk = 1; msk <= 8; msk <<= 1)
      sum += __shfl_xor(sum, msk, 32);
    const float inv = 1.0f / sum;
    const int row = r + half * 8;          // C-layout row within wave tile
#pragma unroll
    for (int t = 0; t < 8; ++t)
      P[wave][row][t * 16 + l16] = (__bf16)(s[t][r] * inv);
  }
  __syncthreads();

  // ---- O = P * V : wave rows x 64 cols (4 tiles), contraction c, 4 steps ----
  v8f o[4];
#pragma unroll
  for (int dt = 0; dt < 4; ++dt) o[dt] = (v8f){};

#pragma unroll
  for (int ks = 0; ks < 4; ++ks) {
    const __bf16* pp = &P[wave][l16][ks * 32 + half * 8];
    v16bf pf = frag_ld(pp, pp + 16);
#pragma unroll
    for (int dt = 0; dt < 4; ++dt) {
      const __bf16* vp = vb + (size_t)(dt * 16 + l16) * BATCH + ks * 32 + half * 8;
      v16bf vf = frag_ld(vp, vp + 16);
      o[dt] = wmma_bf16(pf, vf, o[dt]);
    }
  }

  // store to merged-head layout [b][n][h*64+d], bf16
#pragma unroll
  for (int dt = 0; dt < 4; ++dt) {
    const int d = dt * 16 + l16;
#pragma unroll
    for (int r = 0; r < 8; ++r) {
      const int brow = wave * 16 + r + half * 8;
      out[((size_t)brow * NSAMP + n) * D_MODEL + h * 64 + d] = (__bf16)o[dt][r];
    }
  }
}

// ---------------------------------------------------------------------------
// Output projection: Z[m][e] = sum_d O[m][d]*Wo[e][d] + bo[e], fp32 out.
// A is bf16 row-major -> fragments load straight from global, no LDS.
// ---------------------------------------------------------------------------
__global__ void __launch_bounds__(256)
k_outproj(const __bf16* __restrict__ A, const __bf16* __restrict__ W,
          const float* __restrict__ bias, float* __restrict__ out) {
  const int lane = threadIdx.x & 31, wave = threadIdx.x >> 5;
  const int wm = wave & 1, we = wave >> 1;
  const int half = lane >> 4, l16 = lane & 15;
  const int mBlk = blockIdx.x * 128;
  const int eBlk = blockIdx.y * 128;

  v8f acc[4][2];
#pragma unroll
  for (int mt = 0; mt < 4; ++mt)
#pragma unroll
    for (int et = 0; et < 2; ++et) acc[mt][et] = (v8f){};

  for (int k0 = 0; k0 < D_MODEL; k0 += 32) {
    v16bf bfrag[2];
#pragma unroll
    for (int et = 0; et < 2; ++et) {
      const int e = eBlk + we * 32 + et * 16 + l16;
      const __bf16* bp = W + (size_t)e * D_MODEL + k0 + half * 8;
      bfrag[et] = frag_ld(bp, bp + 16);
    }
#pragma unroll
    for (int mt = 0; mt < 4; ++mt) {
      const int m0 = mBlk + wm * 64 + mt * 16;
      const __bf16* ap = A + (size_t)(m0 + l16) * D_MODEL + k0 + half * 8;
      v16bf afrag = frag_ld(ap, ap + 16);
      acc[mt][0] = wmma_bf16(afrag, bfrag[0], acc[mt][0]);
      acc[mt][1] = wmma_bf16(afrag, bfrag[1], acc[mt][1]);
    }
  }

#pragma unroll
  for (int et = 0; et < 2; ++et) {
    const int e = eBlk + we * 32 + et * 16 + l16;
    const float bv = bias[e];
#pragma unroll
    for (int mt = 0; mt < 4; ++mt) {
#pragma unroll
      for (int r = 0; r < 8; ++r) {
        const int m = mBlk + wm * 64 + mt * 16 + r + half * 8;
        out[(size_t)m * D_MODEL + e] = acc[mt][et][r] + bv;
      }
    }
  }
}

// ---------------------------------------------------------------------------
// Launch
// ---------------------------------------------------------------------------
extern "C" void kernel_launch(void* const* d_in, const int* in_sizes, int n_in,
                              void* d_out, int out_size, void* d_ws, size_t ws_size,
                              hipStream_t stream) {
  (void)in_sizes; (void)n_in; (void)out_size; (void)ws_size;

  const float* query = (const float*)d_in[0];
  const float* key_  = (const float*)d_in[1];
  const float* value = (const float*)d_in[2];
  const float* Wq = (const float*)d_in[3];  const float* bq = (const float*)d_in[4];
  const float* Wk = (const float*)d_in[5];  const float* bk = (const float*)d_in[6];
  const float* Wv = (const float*)d_in[7];  const float* bv = (const float*)d_in[8];
  const float* Wo = (const float*)d_in[9];  const float* bo = (const float*)d_in[10];

  __bf16* ws = (__bf16*)d_ws;
  const size_t WELEM = (size_t)D_MODEL * D_MODEL;           // 1M
  const size_t XELEM = (size_t)M_TOT * D_MODEL;             // 33.5M
  __bf16* wqb = ws;
  __bf16* wkb = wqb + WELEM;
  __bf16* wvb = wkb + WELEM;
  __bf16* wob = wvb + WELEM;
  __bf16* xqb = wob + WELEM;       // bf16 activations
  __bf16* xkb = xqb + XELEM;
  __bf16* xvb = xkb + XELEM;
  __bf16* qws = xvb + XELEM;       // projected q/k/v (attention layouts)
  __bf16* kws = qws + XELEM;
  __bf16* vws = kws + XELEM;
  __bf16* aws = vws + XELEM;       // attention output, [b][n][e] bf16

  // 1) fp32 -> bf16 (weights + activations)
  const int w4 = (int)(WELEM / 4), x4 = (int)(XELEM / 4);
  k_cvt<<<(w4 + 255) / 256, 256, 0, stream>>>(Wq, wqb, w4);
  k_cvt<<<(w4 + 255) / 256, 256, 0, stream>>>(Wk, wkb, w4);
  k_cvt<<<(w4 + 255) / 256, 256, 0, stream>>>(Wv, wvb, w4);
  k_cvt<<<(w4 + 255) / 256, 256, 0, stream>>>(Wo, wob, w4);
  k_cvt<<<(x4 + 255) / 256, 256, 0, stream>>>(query, xqb, x4);
  k_cvt<<<(x4 + 255) / 256, 256, 0, stream>>>(key_,  xkb, x4);
  k_cvt<<<(x4 + 255) / 256, 256, 0, stream>>>(value, xvb, x4);

  // 2) q/k/v projections (v pre-transposed for PV matmul)
  dim3 gproj(M_TOT / 128, D_MODEL / 128, 1);
  k_proj<<<gproj, 256, 0, stream>>>(xqb, wqb, bq, qws, 0);
  k_proj<<<gproj, 256, 0, stream>>>(xkb, wkb, bk, kws, 0);
  k_proj<<<gproj, 256, 0, stream>>>(xvb, wvb, bv, vws, 1);

  // 3) intersample attention, one block per (head, sample)
  k_attn<<<NHEAD * NSAMP, 256, 0, stream>>>(qws, kws, vws, aws);

  // 4) output projection -> fp32
  k_outproj<<<gproj, 256, 0, stream>>>(aws, wob, bo, (float*)d_out);
}
